// ODEDecoderDynamic_14826227106025
// MI455X (gfx1250) — matrix operand order repaired and checked by reading, not compiled, not run on hardware
//
#include <hip/hip_runtime.h>
#include <math.h>

typedef __attribute__((ext_vector_type(16))) _Float16 v16h;
typedef __attribute__((ext_vector_type(8)))  _Float16 v8h;
typedef __attribute__((ext_vector_type(4)))  _Float16 v4h;
typedef __attribute__((ext_vector_type(8)))  float    v8f;

#define N_ROWS 16384
#define D_DIM  64
#define NM     (N_ROWS * D_DIM)   // 1,048,576 elements per (N,D) array
#define NSTEPS 12
#define NSEG   7
#define RTOL_C 0.01f
#define ATOL_C 0.001f

#define CAT16 0,1,2,3,4,5,6,7,8,9,10,11,12,13,14,15

// ---------------------------------------------------------------------------
// ISA 7.12.2 16-bit layouts (wave32):
//  A 16x32 : lane L -> row M=L&15; VGPR0-3 = K(laneHi*8 + 0..7)  [16B contig]
//                                  VGPR4-7 = K(laneHi*8 + 16..23)[16B contig]
//  B 32x16 : lane L -> col N=L&15; same K order as A in ISA; this code uses K(laneHi*16 + 0..15) for A and B
//            -> contiguous 32B iff the tile is stored TRANSPOSED [n][k].
//  C/D 16x16 f32: acc elem r -> (M = r + 8*laneHi, N = L&15)
// ---------------------------------------------------------------------------

// ---------------------------------------------------------------------------
// Fused DOPRI5 stage kernel:
//   ys  = y + h * (c1*p1 + ... + c6*p6)     (first nk terms used)
//   hid = tanh(ys @ W1 + b1)                (f16 WMMA, f32 accumulate)
//   kout = hid @ W2 + b2
// 128 threads = 4 waves; each wave owns a 16-row stripe; WG owns 64 rows.
// ---------------------------------------------------------------------------
__global__ void __launch_bounds__(128)
stage_kernel(const float* __restrict__ y,
             const float* __restrict__ p1, const float* __restrict__ p2,
             const float* __restrict__ p3, const float* __restrict__ p4,
             const float* __restrict__ p5, const float* __restrict__ p6,
             float c1, float c2, float c3, float c4, float c5, float c6,
             int nk,
             const float* __restrict__ st,   // st[2] = h_try
             const float* __restrict__ W1, const float* __restrict__ b1,
             const float* __restrict__ W2, const float* __restrict__ b2,
             float* __restrict__ kout, float* __restrict__ ysout)
{
    __shared__ __align__(16) _Float16 sYs [64 * 64];  // stage input, f16 rows
    __shared__ __align__(16) _Float16 sW1t[64 * 64];  // W1 transposed [n][k]
    __shared__ __align__(16) _Float16 sW2t[64 * 64];  // W2 transposed [n][k]
    __shared__ __align__(16) _Float16 sHid[64 * 64];  // hidden, f16 rows

    const int tid     = threadIdx.x;
    const int rowBase = blockIdx.x * 64;
    const float h     = st[2];

    // Stage weights into LDS transposed (coalesced global reads).
    for (int i = tid; i < 4096; i += 128) {
        int k = i >> 6, n = i & 63;
        sW1t[n * 64 + k] = (_Float16)W1[i];
        sW2t[n * 64 + k] = (_Float16)W2[i];
    }

    // Fused stage combination, float4-vectorized, f16 store to LDS.
    {
        const float4* yv  = (const float4*)y;
        const float4* q1  = (const float4*)p1;
        const float4* q2  = (const float4*)p2;
        const float4* q3  = (const float4*)p3;
        const float4* q4  = (const float4*)p4;
        const float4* q5  = (const float4*)p5;
        const float4* q6  = (const float4*)p6;
        float4* ysv = (float4*)ysout;
        v4h* sYv4 = (v4h*)sYs;
        const int base4 = rowBase * 16;            // 16 float4 per row
        for (int i4 = tid; i4 < 1024; i4 += 128) {
            int g4 = base4 + i4;
            float4 v = yv[g4];
            float4 a = make_float4(0.f, 0.f, 0.f, 0.f);
            if (nk > 0) { float4 q = q1[g4]; a.x += c1*q.x; a.y += c1*q.y; a.z += c1*q.z; a.w += c1*q.w; }
            if (nk > 1) { float4 q = q2[g4]; a.x += c2*q.x; a.y += c2*q.y; a.z += c2*q.z; a.w += c2*q.w; }
            if (nk > 2) { float4 q = q3[g4]; a.x += c3*q.x; a.y += c3*q.y; a.z += c3*q.z; a.w += c3*q.w; }
            if (nk > 3) { float4 q = q4[g4]; a.x += c4*q.x; a.y += c4*q.y; a.z += c4*q.z; a.w += c4*q.w; }
            if (nk > 4) { float4 q = q5[g4]; a.x += c5*q.x; a.y += c5*q.y; a.z += c5*q.z; a.w += c5*q.w; }
            if (nk > 5) { float4 q = q6[g4]; a.x += c6*q.x; a.y += c6*q.y; a.z += c6*q.z; a.w += c6*q.w; }
            v.x += h * a.x; v.y += h * a.y; v.z += h * a.z; v.w += h * a.w;
            v4h hv;
            hv[0] = (_Float16)v.x; hv[1] = (_Float16)v.y;
            hv[2] = (_Float16)v.z; hv[3] = (_Float16)v.w;
            sYv4[i4] = hv;
            if (ysout) ysv[g4] = v;
        }
    }
    __syncthreads();

    const int lane   = tid & 31;
    const int wave   = tid >> 5;
    const int laneHi = lane >> 4;
    const int ln     = lane & 15;
    const int mrow   = wave * 16;            // this wave's local row stripe

    const v8h* sYv  = (const v8h*)sYs;       // units of 8 halves (16 B)
    const v8h* sHv  = (const v8h*)sHid;
    const v8h* bW1  = (const v8h*)sW1t;
    const v8h* bW2  = (const v8h*)sW2t;
    const int  arow = (mrow + ln) * 8;       // row * 64 halves / 8

    // ---------------- GEMM 1: hid = tanh(ys @ W1 + b1) ----------------
    v8f acc[4] = {};
    #pragma unroll
    for (int kt = 0; kt < 2; ++kt) {         // kk = kt*32
        v8h alo = sYv[arow + kt * 4 + laneHi * 2];
        v8h ahi = sYv[arow + kt * 4 + laneHi * 2 + 1];
        v16h a = __builtin_shufflevector(alo, ahi, CAT16);
        #pragma unroll
        for (int ct = 0; ct < 4; ++ct) {
            int bi = (ct * 16 + ln) * 8 + kt * 4 + laneHi * 2;
            v16h b = __builtin_shufflevector(bW1[bi], bW1[bi + 1], CAT16);
            acc[ct] = __builtin_amdgcn_wmma_f32_16x16x32_f16(
                false, a, false, b, (short)0, acc[ct], false, false);
        }
    }
    #pragma unroll
    for (int ct = 0; ct < 4; ++ct) {
        #pragma unroll
        for (int r = 0; r < 8; ++r) {
            int row = mrow + r + laneHi * 8;
            int col = ct * 16 + ln;
            sHid[row * 64 + col] = (_Float16)tanhf(acc[ct][r] + b1[col]);
        }
    }
    __syncthreads();

    // ---------------- GEMM 2: kout = hid @ W2 + b2 ----------------
    v8f acc2[4] = {};
    #pragma unroll
    for (int kt = 0; kt < 2; ++kt) {
        v8h alo = sHv[arow + kt * 4 + laneHi * 2];
        v8h ahi = sHv[arow + kt * 4 + laneHi * 2 + 1];
        v16h a = __builtin_shufflevector(alo, ahi, CAT16);
        #pragma unroll
        for (int ct = 0; ct < 4; ++ct) {
            int bi = (ct * 16 + ln) * 8 + kt * 4 + laneHi * 2;
            v16h b = __builtin_shufflevector(bW2[bi], bW2[bi + 1], CAT16);
            acc2[ct] = __builtin_amdgcn_wmma_f32_16x16x32_f16(
                false, a, false, b, (short)0, acc2[ct], false, false);
        }
    }
    #pragma unroll
    for (int ct = 0; ct < 4; ++ct) {
        #pragma unroll
        for (int r = 0; r < 8; ++r) {
            int row = rowBase + mrow + r + laneHi * 8;
            int col = ct * 16 + ln;
            kout[row * 64 + col] = acc2[ct][r] + b2[col];
        }
    }
}

// ---------------------------------------------------------------------------
// init: ycur = y0; out[0] = y0; h = 0.1*(vt[1]-vt[0])
// ---------------------------------------------------------------------------
__global__ void __launch_bounds__(256)
init_kernel(const float* __restrict__ vt, const float* __restrict__ y0,
            float* __restrict__ ycur, float* __restrict__ out,
            float* __restrict__ st)
{
    int i = blockIdx.x * 256 + threadIdx.x;   // float4 index
    if (i < NM / 4) {
        float4 v = ((const float4*)y0)[i];
        ((float4*)ycur)[i] = v;
        ((float4*)out)[i]  = v;
    }
    if (i == 0) st[1] = (vt[1] - vt[0]) * 0.1f;
}

// st: [0]=t [1]=h [2]=h_try [3]=active [4]=adv
__global__ void pre_kernel(const float* __restrict__ vt, int seg, int step,
                           float* __restrict__ st)
{
    if (threadIdx.x == 0) {
        if (step == 0) st[0] = vt[seg];       // t = t0 at segment start
        float t = st[0], h = st[1];
        float rem = vt[seg + 1] - t;
        st[3] = (rem > 1e-9f) ? 1.f : 0.f;
        st[2] = fminf(h, fmaxf(rem, 1e-9f));
    }
}

// err partial reduction: deterministic fixed-order per-thread + LDS tree.
// 256 blocks x 256 threads, each thread handles 4 float4 = 16 elements.
__global__ void __launch_bounds__(256)
err_kernel(const float* __restrict__ y, const float* __restrict__ y5,
           const float* __restrict__ k1, const float* __restrict__ k3,
           const float* __restrict__ k4, const float* __restrict__ k5,
           const float* __restrict__ k6, const float* __restrict__ k7,
           const float* __restrict__ st, float* __restrict__ partials)
{
    const float E1 = 35.f / 384.f - 5179.f / 57600.f;
    const float E3 = 500.f / 1113.f - 7571.f / 16695.f;
    const float E4 = 125.f / 192.f - 393.f / 640.f;
    const float E5 = -2187.f / 6784.f + 92097.f / 339200.f;
    const float E6 = 11.f / 84.f - 187.f / 2100.f;
    const float E7 = -1.f / 40.f;

    __shared__ float sdata[256];
    const float h = st[2];
    const float4* yv  = (const float4*)y;
    const float4* y5v = (const float4*)y5;
    const float4* v1  = (const float4*)k1;
    const float4* v3  = (const float4*)k3;
    const float4* v4  = (const float4*)k4;
    const float4* v5  = (const float4*)k5;
    const float4* v6  = (const float4*)k6;
    const float4* v7  = (const float4*)k7;

    int base = blockIdx.x * 1024 + threadIdx.x;   // float4 index
    float s = 0.f;
    #pragma unroll
    for (int j = 0; j < 4; ++j) {
        int i = base + j * 256;
        float4 a = v1[i], b = v3[i], c = v4[i], d = v5[i], e = v6[i], f = v7[i];
        float4 yy = yv[i], zz = y5v[i];
        float ex, sc, r;
        ex = h * (E1*a.x + E3*b.x + E4*c.x + E5*d.x + E6*e.x + E7*f.x);
        sc = ATOL_C + RTOL_C * fmaxf(fabsf(yy.x), fabsf(zz.x)); r = ex / sc; s += r * r;
        ex = h * (E1*a.y + E3*b.y + E4*c.y + E5*d.y + E6*e.y + E7*f.y);
        sc = ATOL_C + RTOL_C * fmaxf(fabsf(yy.y), fabsf(zz.y)); r = ex / sc; s += r * r;
        ex = h * (E1*a.z + E3*b.z + E4*c.z + E5*d.z + E6*e.z + E7*f.z);
        sc = ATOL_C + RTOL_C * fmaxf(fabsf(yy.z), fabsf(zz.z)); r = ex / sc; s += r * r;
        ex = h * (E1*a.w + E3*b.w + E4*c.w + E5*d.w + E6*e.w + E7*f.w);
        sc = ATOL_C + RTOL_C * fmaxf(fabsf(yy.w), fabsf(zz.w)); r = ex / sc; s += r * r;
    }
    sdata[threadIdx.x] = s;
    __syncthreads();
    for (int off = 128; off > 0; off >>= 1) {
        if (threadIdx.x < off) sdata[threadIdx.x] += sdata[threadIdx.x + off];
        __syncthreads();
    }
    if (threadIdx.x == 0) partials[blockIdx.x] = sdata[0];
}

// finalize: global err_norm + step controller update (single workgroup)
__global__ void __launch_bounds__(256)
finalize_kernel(const float* __restrict__ partials, int nparts,
                float* __restrict__ st)
{
    __shared__ float sdata[256];
    float s = 0.f;
    for (int i = threadIdx.x; i < nparts; i += 256) s += partials[i];
    sdata[threadIdx.x] = s;
    __syncthreads();
    for (int off = 128; off > 0; off >>= 1) {
        if (threadIdx.x < off) sdata[threadIdx.x] += sdata[threadIdx.x + off];
        __syncthreads();
    }
    if (threadIdx.x == 0) {
        float err_norm = sqrtf(sdata[0] / (float)NM);
        err_norm = fmaxf(err_norm, 1e-10f);
        float factor = 0.9f * powf(err_norm, -0.2f);
        factor = fminf(fmaxf(factor, 0.2f), 10.f);
        bool  accept = (err_norm <= 1.f);
        float active = st[3];
        float h_try  = st[2];
        bool  adv    = accept && (active > 0.5f);
        if (adv) st[0] = st[0] + h_try;
        if (active > 0.5f) st[1] = h_try * factor;
        st[4] = adv ? 1.f : 0.f;
    }
}

__global__ void __launch_bounds__(256)
update_kernel(float* __restrict__ y, const float* __restrict__ y5,
              const float* __restrict__ st)
{
    int i = blockIdx.x * 256 + threadIdx.x;   // float4 index
    if (i < NM / 4) {
        if (st[4] > 0.5f) ((float4*)y)[i] = ((const float4*)y5)[i];
    }
}

__global__ void __launch_bounds__(256)
copy_kernel(const float* __restrict__ src, float* __restrict__ dst)
{
    int i = blockIdx.x * 256 + threadIdx.x;   // float4 index
    if (i < NM / 4) ((float4*)dst)[i] = ((const float4*)src)[i];
}

// ---------------------------------------------------------------------------
// Host launcher: fixed 7 segments x 12 steps graph of kernels.
// d_ws layout (floats): ycur | y5 | k1..k7 | partials(256) | st(8)
//   -> 9*NM + 264 floats ~= 37.8 MB of scratch.
// ---------------------------------------------------------------------------
extern "C" void kernel_launch(void* const* d_in, const int* in_sizes, int n_in,
                              void* d_out, int out_size, void* d_ws, size_t ws_size,
                              hipStream_t stream)
{
    const float* vt = (const float*)d_in[0];
    const float* y0 = (const float*)d_in[1];
    const float* W1 = (const float*)d_in[2];
    const float* b1 = (const float*)d_in[3];
    const float* W2 = (const float*)d_in[4];
    const float* b2 = (const float*)d_in[5];
    float* out = (float*)d_out;

    float* ws   = (float*)d_ws;
    float* ycur = ws;
    float* y5   = ws + (size_t)NM;
    float* kb[7];
    for (int i = 0; i < 7; ++i) kb[i] = ws + (size_t)NM * (2 + i);
    float* partials = ws + (size_t)NM * 9;
    float* st       = partials + 256;

    const dim3 gblk(128);
    const dim3 ggrid(N_ROWS / 64);        // 256 workgroups
    const int  egrid = 256;               // err partial blocks
    const int  cgrid = NM / 4 / 256;      // 1024 blocks for float4 elementwise

    init_kernel<<<cgrid, 256, 0, stream>>>(vt, y0, ycur, out, st);

    for (int seg = 0; seg < NSEG; ++seg) {
        for (int step = 0; step < NSTEPS; ++step) {
            pre_kernel<<<1, 32, 0, stream>>>(vt, seg, step, st);

            // k1 = f(y)
            stage_kernel<<<ggrid, gblk, 0, stream>>>(ycur,
                ycur, ycur, ycur, ycur, ycur, ycur,
                0.f, 0.f, 0.f, 0.f, 0.f, 0.f, 0,
                st, W1, b1, W2, b2, kb[0], nullptr);
            // k2 = f(y + h*0.2*k1)
            stage_kernel<<<ggrid, gblk, 0, stream>>>(ycur,
                kb[0], ycur, ycur, ycur, ycur, ycur,
                0.2f, 0.f, 0.f, 0.f, 0.f, 0.f, 1,
                st, W1, b1, W2, b2, kb[1], nullptr);
            // k3
            stage_kernel<<<ggrid, gblk, 0, stream>>>(ycur,
                kb[0], kb[1], ycur, ycur, ycur, ycur,
                3.f/40.f, 9.f/40.f, 0.f, 0.f, 0.f, 0.f, 2,
                st, W1, b1, W2, b2, kb[2], nullptr);
            // k4
            stage_kernel<<<ggrid, gblk, 0, stream>>>(ycur,
                kb[0], kb[1], kb[2], ycur, ycur, ycur,
                44.f/45.f, -56.f/15.f, 32.f/9.f, 0.f, 0.f, 0.f, 3,
                st, W1, b1, W2, b2, kb[3], nullptr);
            // k5
            stage_kernel<<<ggrid, gblk, 0, stream>>>(ycur,
                kb[0], kb[1], kb[2], kb[3], ycur, ycur,
                19372.f/6561.f, -25360.f/2187.f, 64448.f/6561.f, -212.f/729.f, 0.f, 0.f, 4,
                st, W1, b1, W2, b2, kb[4], nullptr);
            // k6
            stage_kernel<<<ggrid, gblk, 0, stream>>>(ycur,
                kb[0], kb[1], kb[2], kb[3], kb[4], ycur,
                9017.f/3168.f, -355.f/33.f, 46732.f/5247.f, 49.f/176.f, -5103.f/18656.f, 0.f, 5,
                st, W1, b1, W2, b2, kb[5], nullptr);
            // y5 = y + h*(35/384 k1 + 500/1113 k3 + 125/192 k4 - 2187/6784 k5 + 11/84 k6); k7 = f(y5)
            stage_kernel<<<ggrid, gblk, 0, stream>>>(ycur,
                kb[0], kb[2], kb[3], kb[4], kb[5], ycur,
                35.f/384.f, 500.f/1113.f, 125.f/192.f, -2187.f/6784.f, 11.f/84.f, 0.f, 5,
                st, W1, b1, W2, b2, kb[6], y5);

            err_kernel<<<egrid, 256, 0, stream>>>(ycur, y5,
                kb[0], kb[2], kb[3], kb[4], kb[5], kb[6], st, partials);
            finalize_kernel<<<1, 256, 0, stream>>>(partials, egrid, st);
            update_kernel<<<cgrid, 256, 0, stream>>>(ycur, y5, st);
        }
        copy_kernel<<<cgrid, 256, 0, stream>>>(ycur, out + (size_t)(seg + 1) * NM);
    }
}
